// DynamicRuleEmbedding_38878043963938
// MI455X (gfx1250) — matrix operand order repaired and checked by reading, not compiled, not run on hardware
//
#include <hip/hip_runtime.h>
#include <hip/hip_bf16.h>
#include <math.h>

typedef __attribute__((ext_vector_type(8)))  __bf16 v8bf;
typedef __attribute__((ext_vector_type(16))) __bf16 v16bf;
typedef __attribute__((ext_vector_type(8)))  float  v8f;

#define VOCAB   50257
#define EMBED   256
#define NFREQ   64
#define NRULES  100
#define NTOK    4096          // B*S = 4*1024
#define MAXG    360           // >= floor(4096/16) + 100 = 356
#define SORTCAP 8192          // >= 16 * MAXG

// ---------------- K0: init workspace control structures ----------------
__global__ void k_init(int* counts, int* fill, int* offsets, int* sorted, int* g2) {
    int i = blockIdx.x * blockDim.x + threadIdx.x;
    if (i < 128) { counts[i] = 0; fill[i] = 0; offsets[i] = 0; }
    if (i < SORTCAP) sorted[i] = -1;
    if (i < 2 * MAXG) g2[i] = -1;
}

// ---------------- K1: fourier features + projection -> base (bf16), count rules ----
// one block (64 threads) per token
__global__ void k_base(const int* __restrict__ token_ids,
                       const float* __restrict__ a_n,
                       const float* __restrict__ b_n,
                       const int* __restrict__ token_rules,
                       const float* __restrict__ proj_w,
                       const float* __restrict__ proj_b,
                       __bf16* __restrict__ baseb,
                       int* __restrict__ rule_of_token,
                       int* __restrict__ counts) {
    __shared__ float four[NFREQ];
    const int t = blockIdx.x;
    const int f = threadIdx.x;            // 0..63
    const int tok = token_ids[t];

    const float x   = (float)tok * (1.0f / (float)VOCAB);
    const float ang = 6.283185307179586f * (float)(f + 1) * x;
    float sn, cs;
    __sincosf(ang, &sn, &cs);
    four[f] = a_n[(size_t)tok * NFREQ + f] * cs + b_n[(size_t)tok * NFREQ + f] * sn;

    if (f == 0) {
        int r = token_rules[tok];
        rule_of_token[t] = r;
        atomicAdd(&counts[r], 1);
    }
    __syncthreads();

    #pragma unroll
    for (int i = 0; i < 4; ++i) {
        const int d = f + 64 * i;         // 256 outputs over 64 threads
        float acc = proj_b[d];
        const float* wrow = proj_w + (size_t)d * NFREQ;
        #pragma unroll
        for (int ff = 0; ff < NFREQ; ++ff) acc += four[ff] * wrow[ff];
        baseb[(size_t)t * EMBED + d] = (__bf16)acc;
    }
}

// ---------------- Kt: rule_transform [r][d][e] f32 -> Rt [r][e][d] bf16 ------------
__global__ void k_transpose(const float* __restrict__ rt, __bf16* __restrict__ Rt) {
    __shared__ float tile[32][33];
    const int r  = blockIdx.z;
    const int d0 = blockIdx.x * 32;
    const int e0 = blockIdx.y * 32;
    const int tx = threadIdx.x, ty = threadIdx.y;
    tile[ty][tx] = rt[(size_t)r * EMBED * EMBED + (size_t)(d0 + ty) * EMBED + (e0 + tx)];
    __syncthreads();
    Rt[(size_t)r * EMBED * EMBED + (size_t)(e0 + ty) * EMBED + (d0 + tx)] = (__bf16)tile[tx][ty];
}

// ---------------- K2: serial scan -> 16-aligned offsets + group table --------------
__global__ void k_scan(const int* __restrict__ counts, int* __restrict__ offsets,
                       int* __restrict__ g2) {
    if (threadIdx.x != 0 || blockIdx.x != 0) return;
    int off = 0, g = 0;
    for (int r = 0; r < NRULES; ++r) {
        offsets[r] = off;
        const int c  = counts[r];
        const int ng = (c + 15) >> 4;
        for (int j = 0; j < ng; ++j) {
            g2[2 * g + 0] = r;               // rule of this group
            g2[2 * g + 1] = off + 16 * j;    // row base in sorted[]
            ++g;
        }
        off += 16 * ng;
    }
}

// ---------------- K3: scatter token ids into rule buckets --------------------------
__global__ void k_scatter(const int* __restrict__ rule_of_token,
                          const int* __restrict__ offsets,
                          int* __restrict__ fill, int* __restrict__ sorted) {
    const int t = blockIdx.x * blockDim.x + threadIdx.x;
    if (t >= NTOK) return;
    const int r = rule_of_token[t];
    const int pos = offsets[r] + atomicAdd(&fill[r], 1);
    sorted[pos] = t;
}

// ---------------- K4: grouped GEMM, 16 tokens x (256x256) rule, bf16 WMMA ----------
// 512 threads = 16 waves; wave w computes the 16x16 output tile at columns [16w,16w+16)
__global__ void __launch_bounds__(512)
k_wmma(const int* __restrict__ g2, const int* __restrict__ sorted,
       const __bf16* __restrict__ baseb, const __bf16* __restrict__ Rt,
       float* __restrict__ out) {
    const int rule = g2[2 * blockIdx.x + 0];
    if (rule < 0) return;                      // unused group slot (block-uniform)
    const int rowbase = g2[2 * blockIdx.x + 1];

    __shared__ __bf16 shA[16][EMBED];          // 16 tokens x 256 features, 8 KB

    // cooperative A-tile fill: each thread moves 8 contiguous bf16 (16B)
    {
        const int r   = threadIdx.x >> 5;      // 0..15 token row in group
        const int col = (threadIdx.x & 31) * 8;
        const int t   = sorted[rowbase + r];
        v8bf v;
        if (t >= 0) {
            v = *reinterpret_cast<const v8bf*>(baseb + (size_t)t * EMBED + col);
        } else {
            #pragma unroll
            for (int i = 0; i < 8; ++i) v[i] = (__bf16)0.0f;
        }
        *reinterpret_cast<v8bf*>(&shA[r][col]) = v;
    }
    __syncthreads();

    const int wave = threadIdx.x >> 5;
    const int lane = threadIdx.x & 31;
    const int e0   = wave << 4;                // N tile base
    const int n    = lane & 15;
    const int h    = lane >> 4;

    const __bf16* Brow = Rt + (size_t)rule * EMBED * EMBED + (size_t)(e0 + n) * EMBED;

    union { v16bf v; struct { v8bf lo, hi; } p; } a;
    v8f acc = {};

    #pragma unroll
    for (int s = 0; s < 8; ++s) {              // K = 256 in steps of 32
        // A fragment per ISA layout: lane(m=n) comp c -> k = 32s + (c&7) + 16*(c>>3) + 8h
        a.p.lo = *reinterpret_cast<const v8bf*>(&shA[n][32 * s + 8 * h]);
        a.p.hi = *reinterpret_cast<const v8bf*>(&shA[n][32 * s + 16 + 8 * h]);
        // B fragment per ISA layout: lane(n) comp c -> k = 32s + (c&7) + 16*(c>>3) + 8h
        union { v16bf v; struct { v8bf lo, hi; } p; } bu;
        bu.p.lo = *reinterpret_cast<const v8bf*>(Brow + 32 * s + 8 * h);
        bu.p.hi = *reinterpret_cast<const v8bf*>(Brow + 32 * s + 16 + 8 * h);
        const v16bf b = bu.v;
        acc = __builtin_amdgcn_wmma_f32_16x16x32_bf16(
                  false, a.v, false, b, (short)0, acc, false, false);
    }

    // C/D layout: component v -> m = v + 8h ; lane -> n
    #pragma unroll
    for (int v = 0; v < 8; ++v) {
        const int m = v + 8 * h;
        const int t = sorted[rowbase + m];
        if (t >= 0) out[(size_t)t * EMBED + e0 + n] = acc[v];
    }
}

// ---------------- launch ----------------
extern "C" void kernel_launch(void* const* d_in, const int* in_sizes, int n_in,
                              void* d_out, int out_size, void* d_ws, size_t ws_size,
                              hipStream_t stream) {
    const int*   token_ids   = (const int*)  d_in[0];
    const float* a_n         = (const float*)d_in[1];
    const float* b_n         = (const float*)d_in[2];
    const float* rule_xform  = (const float*)d_in[3];
    const int*   token_rules = (const int*)  d_in[4];
    const float* proj_w      = (const float*)d_in[5];
    const float* proj_b      = (const float*)d_in[6];
    float*       out         = (float*)d_out;

    char* ws = (char*)d_ws;
    __bf16* baseb = (__bf16*)ws;                                  // 4096*256*2 = 2 MiB
    __bf16* Rt    = (__bf16*)(ws + 2097152);                      // 100*256*256*2 = 12.5 MiB
    char* p = ws + 2097152 + 13107200;
    int* counts  = (int*)p; p += 512;
    int* fill    = (int*)p; p += 512;
    int* offsets = (int*)p; p += 512;
    int* ruleof  = (int*)p; p += NTOK * 4;
    int* sorted  = (int*)p; p += SORTCAP * 4;
    int* g2      = (int*)p; p += 2 * MAXG * 4;

    k_init<<<40, 256, 0, stream>>>(counts, fill, offsets, sorted, g2);

    k_base<<<NTOK, 64, 0, stream>>>(token_ids, a_n, b_n, token_rules,
                                    proj_w, proj_b, baseb, ruleof, counts);

    k_transpose<<<dim3(8, 8, NRULES), dim3(32, 32), 0, stream>>>(rule_xform, Rt);

    k_scan<<<1, 1, 0, stream>>>(counts, offsets, g2);

    k_scatter<<<(NTOK + 255) / 256, 256, 0, stream>>>(ruleof, offsets, fill, sorted);

    k_wmma<<<MAXG, 512, 0, stream>>>(g2, sorted, baseb, Rt, out);
}
